// NearestNeighborAttention_27590869909790
// MI455X (gfx1250) — matrix operand order repaired and checked by reading, not compiled, not run on hardware
//
#include <hip/hip_runtime.h>
#include <math.h>

// ---------------------------------------------------------------------------
// NearestNeighborAttention for MI455X (gfx1250, wave32)
//   B=1, S=8192, F=256, H=8, D=32, K=32
// d_out = out (S*F floats) ++ metric (H*D = 256 floats)
// d_ws layout (bytes), total ~34.1 MB:
//   [0,        4MB)   x in bf16            (S*F*2)
//   [4MB,  4MB+384KB) Wq|Wk|Wv in bf16     (3*F*F*2)
//   [8MB,     32MB)   q|k|v in fp32        (3*S*F*4)
//   [33MB,    34MB)   nn_idx int32         (S*32*4)
//   [34MB+64KB, 34MB+128KB) metric partials (64*F*4)
// ---------------------------------------------------------------------------

typedef __bf16 bf16;
typedef __attribute__((ext_vector_type(16))) __bf16 v16bf;
typedef __attribute__((ext_vector_type(8)))  __bf16 v8bf;
typedef __attribute__((ext_vector_type(8)))  float  v8f;
typedef __attribute__((ext_vector_type(4)))  int    v4i;

typedef __attribute__((address_space(1))) v4i* global_v4i_p;
typedef __attribute__((address_space(3))) v4i* lds_v4i_p;

union V16BF { v16bf v; v8bf h[2]; };

#define S_PTS 8192
#define F_DIM 256
#define H_CNT 8
#define D_DIM 32
#define K_NBR 32
#define LDS_STRIDE 264   // 256 + 8 bf16 pad -> 528B row stride, avoids 64-bank aliasing

// ---------------------------------------------------------------------------
// Bitonic compare-exchange across lanes on (dist, idx), lexicographic order.
// ---------------------------------------------------------------------------
__device__ __forceinline__ void cmpx_lane(float& d, int& i, int j, bool up) {
    const int lane = threadIdx.x & 31;
    float od = __shfl_xor(d, j, 32);
    int   oi = __shfl_xor(i, j, 32);
    bool lower = (lane & j) == 0;
    bool takeOther = (lower == up) ? (od < d || (od == d && oi < i))
                                   : (od > d || (od == d && oi > i));
    if (takeOther) { d = od; i = oi; }
}

// ---------------------------------------------------------------------------
// KNN: one wave per query. Running top-32 kept sorted across lanes
// (lane 0 = nearest). Per 32-candidate batch: bitonic sort, reverse,
// elementwise-min against running list, 5-stage bitonic cleanup.
// ---------------------------------------------------------------------------
__global__ void nna_knn_kernel(const float* __restrict__ coords,
                               int* __restrict__ nn_idx) {
    const int lane = threadIdx.x & 31;
    const int wave = threadIdx.x >> 5;
    const int s = blockIdx.x * (blockDim.x >> 5) + wave;
    if (s >= S_PTS) return;

    const float qx = coords[3 * s + 0];
    const float qy = coords[3 * s + 1];
    const float qz = coords[3 * s + 2];

    float rd = __builtin_inff();   // running dist, sorted ascending across lanes
    int   ri = -1;

    for (int base = 0; base < S_PTS; base += 32) {
        const int p = base + lane;
        float dx = coords[3 * p + 0] - qx;
        float dy = coords[3 * p + 1] - qy;
        float dz = coords[3 * p + 2] - qz;
        float d  = dx * dx + dy * dy + dz * dz;
        if (p == s) d = __builtin_inff();   // exclude self (rank-0 in argsort)

        // skip batch if nothing beats current worst of the running top-32
        float runMax = __shfl(rd, 31, 32);
        if (__ballot(d < runMax) == 0ull) continue;

        // bitonic sort the 32 candidates ascending across lanes
        float cd = d; int ci = p;
        #pragma unroll
        for (int k = 2; k <= 32; k <<= 1) {
            #pragma unroll
            for (int j = k >> 1; j > 0; j >>= 1)
                cmpx_lane(cd, ci, j, (lane & k) == 0);
        }
        // reverse -> descending; pairwise min with ascending running list
        float bd = __shfl_xor(cd, 31, 32);
        int   bi = __shfl_xor(ci, 31, 32);
        if (bd < rd || (bd == rd && bi < ri)) { rd = bd; ri = bi; }
        // cleanup: bitonic merge (ascending)
        #pragma unroll
        for (int j = 16; j > 0; j >>= 1)
            cmpx_lane(rd, ri, j, true);
    }

    nn_idx[s * K_NBR + lane] = ri;   // lane k = k-th nearest neighbor
}

// ---------------------------------------------------------------------------
// fp32 -> bf16 conversion (x, Wq, Wk, Wv)
// ---------------------------------------------------------------------------
__global__ void nna_cvt_bf16_kernel(const float* __restrict__ in,
                                    bf16* __restrict__ out, int n) {
    int i = blockIdx.x * blockDim.x + threadIdx.x;
    if (i < n) out[i] = (bf16)in[i];
}

// ---------------------------------------------------------------------------
// Fused Q/K/V projection: out[which] = x @ W[which]^T via
// v_wmma_f32_16x16x32_bf16.
// Block = 8 waves, one (which, mt) strip: the 16x256 A strip (8 KB bf16) is
// async-staged into LDS once (global_load_async_to_lds_b128 -> ASYNCcnt),
// then each wave computes TWO 16x16 N-tiles, reusing the A fragment.
// A (16x32 bf16): lane<16 -> row=lane, k in {0..7, 16..23};
//                 lane>=16 -> row=lane-16, k in {8..15, 24..31}.
// B (32x16 bf16): lane<16 -> col=lane, k in {0..7, 16..23}; lane>=16 -> col=lane-16, k in {8..15, 24..31}.
//                 B[k][n] = W^T[k][n] = W[n][k] -> row-major rows of W.
// ---------------------------------------------------------------------------
__global__ void nna_qkv_gemm_kernel(const bf16* __restrict__ xb,
                                    const bf16* __restrict__ wb,   // [3][F][F]
                                    float* __restrict__ qkv) {     // [3][S][F]
    __shared__ bf16 aTile[16 * LDS_STRIDE];   // padded 16x256 strip

    const int lane  = threadIdx.x & 31;
    const int wave  = threadIdx.x >> 5;
    const int which = blockIdx.x >> 9;        // 0..2  (blocks = 3*512)
    const int mt    = blockIdx.x & 511;       // M tile index

    // ---- async-stage A strip: 512 chunks of 16B, 2 per thread ----
    {
        const int t = threadIdx.x;
        #pragma unroll
        for (int c = t; c < 512; c += 256) {
            const int row = c >> 5;                // 0..15
            const int col = (c & 31) << 3;         // element col, 16B steps
            const bf16* g = xb + (size_t)(mt * 16 + row) * F_DIM + col;
            bf16* l = &aTile[row * LDS_STRIDE + col];
            __builtin_amdgcn_global_load_async_to_lds_b128(
                (global_v4i_p)(g), (lds_v4i_p)(l),
                /*offset=*/0, /*cpol=*/0);
        }
    }
    __builtin_amdgcn_s_wait_asynccnt(0);
    __syncthreads();

    const bf16* W   = wb  + (size_t)which * F_DIM * F_DIM;
    float*      out = qkv + (size_t)which * S_PTS * F_DIM;

    const int aoff = (lane < 16) ? 0 : 8;     // A k-offset within 32-block
    const int boff = (lane < 16) ? 0 : 8;     // B k-offset within 32-block
    const int nt0  = wave * 2;                // this wave's two N tiles

    const bf16* aRow   = &aTile[(lane & 15) * LDS_STRIDE];
    const bf16* b0Base = W + (size_t)(nt0 * 16 + (lane & 15)) * F_DIM;
    const bf16* b1Base = b0Base + 16 * F_DIM;

    v8f acc0 = {}, acc1 = {};
    #pragma unroll
    for (int kb = 0; kb < F_DIM; kb += 32) {
        V16BF A, B0, B1;
        A.h[0]  = *(const v8bf*)(aRow + kb + aoff);
        A.h[1]  = *(const v8bf*)(aRow + kb + aoff + 16);
        B0.h[0] = *(const v8bf*)(b0Base + kb + boff);
        B0.h[1] = *(const v8bf*)(b0Base + kb + boff + 16);
        B1.h[0] = *(const v8bf*)(b1Base + kb + boff);
        B1.h[1] = *(const v8bf*)(b1Base + kb + boff + 16);
        acc0 = __builtin_amdgcn_wmma_f32_16x16x32_bf16(
                   false, A.v, false, B0.v, (short)0, acc0, false, false);
        acc1 = __builtin_amdgcn_wmma_f32_16x16x32_bf16(
                   false, A.v, false, B1.v, (short)0, acc1, false, false);
    }

    // D layout: VGPR r -> M = r + 8*(lane>=16), N = lane&15
    const int mbase = mt * 16 + ((lane >> 4) << 3);
    const int ncol  = lane & 15;
    #pragma unroll
    for (int r = 0; r < 8; r++) {
        out[(size_t)(mbase + r) * F_DIM + nt0 * 16 + ncol]       = acc0[r];
        out[(size_t)(mbase + r) * F_DIM + (nt0 + 1) * 16 + ncol] = acc1[r];
    }
}

// ---------------------------------------------------------------------------
// Attention over the 32 gathered neighbors. One wave per (s, h).
// Lane k owns neighbor k for the score phase; lane d owns output dim d
// for the accumulation phase.
// ---------------------------------------------------------------------------
__global__ void nna_attn_kernel(const float* __restrict__ q,
                                const float* __restrict__ kmat,
                                const float* __restrict__ vmat,
                                const int* __restrict__ nn_idx,
                                float* __restrict__ out) {
    const int lane = threadIdx.x & 31;
    const int wave = threadIdx.x >> 5;
    const int g = blockIdx.x * (blockDim.x >> 5) + wave;   // 0 .. S*H-1
    const int s = g >> 3;                                  // H = 8
    const int h = g & 7;
    if (s >= S_PTS) return;

    const int nbr = nn_idx[s * K_NBR + lane];
    const float* qp = q    + (size_t)s   * F_DIM + h * D_DIM;
    const float* kp = kmat + (size_t)nbr * F_DIM + h * D_DIM;

    float score = 0.f;
    #pragma unroll
    for (int d = 0; d < D_DIM; d += 4) {
        float4 qq = *(const float4*)(qp + d);
        float4 kk = *(const float4*)(kp + d);
        score += qq.x * kk.x + qq.y * kk.y + qq.z * kk.z + qq.w * kk.w;
    }
    score *= 0.17677669529663687f;   // 1/sqrt(32)

    float m = score;
    #pragma unroll
    for (int j = 16; j > 0; j >>= 1) m = fmaxf(m, __shfl_xor(m, j, 32));
    float e = __expf(score - m);
    float sum = e;
    #pragma unroll
    for (int j = 16; j > 0; j >>= 1) sum += __shfl_xor(sum, j, 32);
    const float attn = e / sum;

    // lane d: out[d] = sum_k attn_k * v[nbr_k][h*32 + d]  (coalesced loads)
    float acc = 0.f;
    #pragma unroll
    for (int k2 = 0; k2 < K_NBR; k2++) {
        float ak = __shfl(attn, k2, 32);
        int   nk = __shfl(nbr,  k2, 32);
        acc += ak * vmat[(size_t)nk * F_DIM + h * D_DIM + lane];
    }
    out[(size_t)s * F_DIM + h * D_DIM + lane] = acc;
}

// ---------------------------------------------------------------------------
// metric = mean_s k[s, :]  -> two-stage column reduction
// ---------------------------------------------------------------------------
__global__ void nna_kmean_partial_kernel(const float* __restrict__ kmat,
                                         float* __restrict__ partial) {
    const int f = threadIdx.x;           // 256 threads
    const int b = blockIdx.x;            // 64 blocks, 128 rows each
    float acc = 0.f;
    const int r0 = b * (S_PTS / 64);
    for (int r = r0; r < r0 + (S_PTS / 64); r++)
        acc += kmat[(size_t)r * F_DIM + f];
    partial[b * F_DIM + f] = acc;
}

__global__ void nna_kmean_final_kernel(const float* __restrict__ partial,
                                       float* __restrict__ metric) {
    const int f = threadIdx.x;
    float acc = 0.f;
    #pragma unroll
    for (int b = 0; b < 64; b++) acc += partial[b * F_DIM + f];
    metric[f] = acc * (1.0f / S_PTS);
}

// ---------------------------------------------------------------------------
extern "C" void kernel_launch(void* const* d_in, const int* in_sizes, int n_in,
                              void* d_out, int out_size, void* d_ws, size_t ws_size,
                              hipStream_t stream) {
    const float* x      = (const float*)d_in[0];
    const float* coords = (const float*)d_in[1];
    const float* Wq     = (const float*)d_in[2];
    const float* Wk     = (const float*)d_in[3];
    const float* Wv     = (const float*)d_in[4];

    char* ws = (char*)d_ws;
    bf16*  xb      = (bf16*)(ws);                                   // 4 MB
    bf16*  wb      = (bf16*)(ws + (size_t)(4 << 20));               // 384 KB
    float* qkv     = (float*)(ws + (size_t)(8 << 20));              // 24 MB
    int*   nn      = (int*)  (ws + (size_t)(32 << 20) + (1 << 20)); // @33MB
    float* partial = (float*)(ws + (size_t)(33 << 20) + (1 << 20) + (1 << 16));

    float* out    = (float*)d_out;
    float* metric = out + (size_t)S_PTS * F_DIM;

    // 1) neighbor graph
    nna_knn_kernel<<<S_PTS / 8, 256, 0, stream>>>(coords, nn);

    // 2) bf16 staging for WMMA
    nna_cvt_bf16_kernel<<<(S_PTS * F_DIM + 255) / 256, 256, 0, stream>>>(x,  xb, S_PTS * F_DIM);
    nna_cvt_bf16_kernel<<<(F_DIM * F_DIM + 255) / 256, 256, 0, stream>>>(Wq, wb,                     F_DIM * F_DIM);
    nna_cvt_bf16_kernel<<<(F_DIM * F_DIM + 255) / 256, 256, 0, stream>>>(Wk, wb +     F_DIM * F_DIM, F_DIM * F_DIM);
    nna_cvt_bf16_kernel<<<(F_DIM * F_DIM + 255) / 256, 256, 0, stream>>>(Wv, wb + 2 * F_DIM * F_DIM, F_DIM * F_DIM);

    // 3) fused q/k/v GEMMs (WMMA + async LDS staging); 3*512 blocks
    nna_qkv_gemm_kernel<<<3 * (S_PTS / 16), 256, 0, stream>>>(xb, wb, qkv);

    // 4) gather + attention
    nna_attn_kernel<<<(S_PTS * H_CNT) / 8, 256, 0, stream>>>(
        qkv, qkv + (size_t)S_PTS * F_DIM, qkv + 2 * (size_t)S_PTS * F_DIM, nn, out);

    // 5) metric = k.mean(axis=1)
    nna_kmean_partial_kernel<<<64, 256, 0, stream>>>(qkv + (size_t)S_PTS * F_DIM, partial);
    nna_kmean_final_kernel<<<1, 256, 0, stream>>>(partial, metric);
}
